// graph_constructor_89275190215363
// MI455X (gfx1250) — matrix operand order repaired and checked by reading, not compiled, not run on hardware
//
#include <hip/hip_runtime.h>
#include <hip/hip_bf16.h>

#define NNODES 8192
#define DIM    64
#define ALPHA  3.0f

typedef __attribute__((ext_vector_type(16))) __bf16 v16bf;
typedef __attribute__((ext_vector_type(8)))  float  v8f;

__device__ __forceinline__ unsigned short f2bf(float f) {
    unsigned u = __float_as_uint(f);
    u += 0x7FFFu + ((u >> 16) & 1u);   // round-to-nearest-even
    return (unsigned short)(u >> 16);
}

// gfx1250 has a native V_TANH_F32 transcendental; use it when clang exposes it.
__device__ __forceinline__ float fast_tanh(float x) {
#if __has_builtin(__builtin_amdgcn_tanhf)
    return __builtin_amdgcn_tanhf(x);
#else
    return tanhf(x);
#endif
}

// ---------------------------------------------------------------------------
// Kernel 1: nodevec = tanh(ALPHA * (emb[idx] @ W^T + b)), stored as bf16.
// 256 threads = 4 nodes x 64 dims per block.
// ---------------------------------------------------------------------------
__global__ void proj_kernel(const int* __restrict__ idx,
                            const float* __restrict__ emb1,
                            const float* __restrict__ emb2,
                            const float* __restrict__ w1,
                            const float* __restrict__ b1,
                            const float* __restrict__ w2,
                            const float* __restrict__ b2,
                            unsigned short* __restrict__ nv1,
                            unsigned short* __restrict__ nv2) {
    __shared__ float se1[4][DIM];
    __shared__ float se2[4][DIM];
    const int local = threadIdx.x / DIM;      // 0..3
    const int d     = threadIdx.x % DIM;
    const int node  = blockIdx.x * 4 + local;
    const int src   = idx[node];
    se1[local][d] = emb1[(size_t)src * DIM + d];
    se2[local][d] = emb2[(size_t)src * DIM + d];
    __syncthreads();
    float acc1 = 0.f, acc2 = 0.f;
    const float* w1r = w1 + (size_t)d * DIM;  // out[d] = sum_c emb[c]*W[d][c]
    const float* w2r = w2 + (size_t)d * DIM;
    #pragma unroll 8
    for (int c = 0; c < DIM; ++c) {
        acc1 += se1[local][c] * w1r[c];
        acc2 += se2[local][c] * w2r[c];
    }
    nv1[(size_t)node * DIM + d] = f2bf(fast_tanh(ALPHA * (acc1 + b1[d])));
    nv2[(size_t)node * DIM + d] = f2bf(fast_tanh(ALPHA * (acc2 + b2[d])));
}

// ---------------------------------------------------------------------------
// WMMA fragment loaders (CDNA5 ISA 7.12.2 bf16 layouts, wave32).
// A (16x32, MxK): lane = half*16+m holds, as 16 bf16:
//   elems 0..7  -> K = kbase + half*8      .. +8   (row i0+m)
//   elems 8..15 -> K = kbase + 16 + half*8 .. +8
// B (32x16, KxN): lane = khalf*16+n holds, for column n (row j0+n of source):
//   elems 0..7  -> K = kbase + khalf*8      .. +8
//   elems 8..15 -> K = kbase + 16 + khalf*8 .. +8
// ---------------------------------------------------------------------------
__device__ __forceinline__ v16bf load_a_frag(const unsigned short* __restrict__ base,
                                             int row0, int kbase, int lane) {
    const int m = lane & 15, half = lane >> 4;
    const unsigned short* p = base + (size_t)(row0 + m) * DIM + kbase + half * 8;
    union { uint4 u[2]; v16bf v; } f;
    f.u[0] = *(const uint4*)(p);        // 8 bf16 = 16B
    f.u[1] = *(const uint4*)(p + 16);   // next K group (+16 elements)
    return f.v;
}

__device__ __forceinline__ v16bf load_b_frag(const unsigned short* __restrict__ base,
                                             int col0, int kbase, int lane) {
    const int n = lane & 15, khalf = lane >> 4;
    const unsigned short* p = base + (size_t)(col0 + n) * DIM + kbase + khalf * 8;
    union { uint4 u[2]; v16bf v; } f;
    f.u[0] = *(const uint4*)(p);        // 8 bf16 = 16B
    f.u[1] = *(const uint4*)(p + 16);   // next K group (+16 elements)
    return f.v;
}

// ---------------------------------------------------------------------------
// Kernel 2: adj = relu(tanh(ALPHA * (V1 V2^T - V2 V1^T))) via bf16 WMMA.
// 8 waves/block, each wave owns a 16x16 tile; block tile = 32 rows x 64 cols.
// ---------------------------------------------------------------------------
__global__ void gemm_adj_kernel(const unsigned short* __restrict__ nv1,
                                const unsigned short* __restrict__ nv2,
                                float* __restrict__ out) {
    const int lane = threadIdx.x & 31;
    const int wave = threadIdx.x >> 5;                    // 0..7
    const int i0 = blockIdx.y * 32 + (wave >> 2) * 16;    // output row base
    const int j0 = blockIdx.x * 64 + (wave & 3) * 16;     // output col base

    const v16bf a1k0 = load_a_frag(nv1, i0, 0,  lane);
    const v16bf a1k1 = load_a_frag(nv1, i0, 32, lane);
    const v16bf a2k0 = load_a_frag(nv2, i0, 0,  lane);
    const v16bf a2k1 = load_a_frag(nv2, i0, 32, lane);
    const v16bf b2k0 = load_b_frag(nv2, j0, 0,  lane);
    const v16bf b2k1 = load_b_frag(nv2, j0, 32, lane);
    const v16bf b1k0 = load_b_frag(nv1, j0, 0,  lane);
    const v16bf b1k1 = load_b_frag(nv1, j0, 32, lane);

    v8f acc1 = {};
    v8f acc2 = {};
    // s1 = V1 V2^T  (K = 0..31 then 32..63)
    acc1 = __builtin_amdgcn_wmma_f32_16x16x32_bf16(false, a1k0, false, b2k0,
                                                   (short)0, acc1, false, false);
    acc1 = __builtin_amdgcn_wmma_f32_16x16x32_bf16(false, a1k1, false, b2k1,
                                                   (short)0, acc1, false, false);
    // s2 = V2 V1^T
    acc2 = __builtin_amdgcn_wmma_f32_16x16x32_bf16(false, a2k0, false, b1k0,
                                                   (short)0, acc2, false, false);
    acc2 = __builtin_amdgcn_wmma_f32_16x16x32_bf16(false, a2k1, false, b1k1,
                                                   (short)0, acc2, false, false);

    // C/D layout: lane = half*16+n; VGPR r holds (M = r + 8*half, N = n).
    const int n = lane & 15, half = lane >> 4;
    #pragma unroll
    for (int r = 0; r < 8; ++r) {
        const int m = r + half * 8;
        const float a = acc1[r] - acc2[r];
        float v = fast_tanh(ALPHA * a);
        v = v > 0.f ? v : 0.f;
        out[(size_t)(i0 + m) * NNODES + (j0 + n)] = v;
    }
}

// ---------------------------------------------------------------------------
// Kernel 3: in-place per-row top-k mask + identity.
// One 256-thread block per row; 32 columns per thread held in registers.
// key = adj + 0.01*noise >= 0, so float bits are order-monotonic; binary
// search in bit space for the k-th largest key. Counting uses a wave32
// shuffle reduction, then one LDS atomic per wave.
// ---------------------------------------------------------------------------
__global__ void topk_mask_kernel(const float* __restrict__ noise,
                                 const int* __restrict__ kptr,
                                 float* __restrict__ out) {
    const int row  = blockIdx.x;
    const int tid  = threadIdx.x;                // 0..255
    const int lane = tid & 31;
    const int k    = kptr[0];
    float*       rowp = out   + (size_t)row * NNODES;
    const float* np   = noise + (size_t)row * NNODES;

    float    adjv[32];
    unsigned keyb[32];
    #pragma unroll
    for (int q = 0; q < 32; ++q) {
        const int j = tid + q * 256;
        const float a = rowp[j];
        adjv[q] = a;
        keyb[q] = __float_as_uint(a + 0.01f * np[j]);
    }

    __shared__ int s_cnt;
    unsigned lo = 0u, hi = 0x40000000u;          // keys < 2.0f
    while (lo < hi) {
        const unsigned mid = lo + ((hi - lo + 1u) >> 1);
        if (tid == 0) s_cnt = 0;
        __syncthreads();
        int c = 0;
        #pragma unroll
        for (int q = 0; q < 32; ++q) c += (keyb[q] >= mid) ? 1 : 0;
        // wave32 tree reduction, then one atomic per wave
        #pragma unroll
        for (int off = 16; off > 0; off >>= 1) c += __shfl_xor(c, off, 32);
        if (lane == 0) atomicAdd(&s_cnt, c);
        __syncthreads();
        const int total = s_cnt;
        __syncthreads();                         // protect s_cnt for next iter
        if (total >= k) lo = mid; else hi = mid - 1u;
    }
    const unsigned T = lo;                       // k-th largest key bits

    #pragma unroll
    for (int q = 0; q < 32; ++q) {
        const int j = tid + q * 256;
        float v = (keyb[q] >= T) ? adjv[q] : 0.f;
        if (j == row) v += 1.0f;                 // + eye (adj diag is 0)
        rowp[j] = v;
    }
}

// ---------------------------------------------------------------------------
// Input order: idx, emb1, emb2, lin1_w, lin1_b, lin2_w, lin2_b, noise, k
// ---------------------------------------------------------------------------
extern "C" void kernel_launch(void* const* d_in, const int* in_sizes, int n_in,
                              void* d_out, int out_size, void* d_ws, size_t ws_size,
                              hipStream_t stream) {
    const int*   idx   = (const int*)  d_in[0];
    const float* emb1  = (const float*)d_in[1];
    const float* emb2  = (const float*)d_in[2];
    const float* w1    = (const float*)d_in[3];
    const float* b1    = (const float*)d_in[4];
    const float* w2    = (const float*)d_in[5];
    const float* b2    = (const float*)d_in[6];
    const float* noise = (const float*)d_in[7];
    const int*   kptr  = (const int*)  d_in[8];

    unsigned short* nv1 = (unsigned short*)d_ws;              // [N,64] bf16
    unsigned short* nv2 = nv1 + (size_t)NNODES * DIM;         // [N,64] bf16
    float* out = (float*)d_out;

    // 1) projections -> bf16 nodevecs (2 MB in d_ws)
    proj_kernel<<<NNODES / 4, 256, 0, stream>>>(idx, emb1, emb2, w1, b1, w2, b2,
                                                nv1, nv2);

    // 2) antisymmetric bilinear score + relu(tanh) -> d_out (WMMA)
    dim3 g2(NNODES / 64, NNODES / 32);
    gemm_adj_kernel<<<g2, 256, 0, stream>>>(nv1, nv2, out);

    // 3) per-row top-k mask + identity, in place on d_out
    topk_mask_kernel<<<NNODES, 256, 0, stream>>>(noise, kptr, out);
}
